// ModulateConv2d_39273180954835
// MI455X (gfx1250) — matrix operand order repaired and checked by reading, not compiled, not run on hardware
//
#include <hip/hip_runtime.h>
#include <hip/hip_bf16.h>
#include <stdint.h>

// ---------------------------------------------------------------------------
// ModulateConv2d for MI455X (gfx1250, wave32, WMMA)
//   x:          [16, 512, 1, 4096] f32
//   style:      [16, 512]          f32
//   weight:     [512, 512, 1, 3]   f32
//   lin_weight: [512, 512]         f32
//   bias:       [512]              f32
//   out:        [16, 512, 1, 4096] f32
// ---------------------------------------------------------------------------

#define BATCH 16
#define IN_C  512
#define OUT_C 512
#define W_DIM 512
#define WID   4096

#define LIN_SCALE    0.04419417382415922f    // 1/sqrt(512)
#define WEIGHT_SCALE 0.025515518153991442f   // 1/sqrt(512*3)
#define EPS_DEMOD    1e-8f

typedef __attribute__((ext_vector_type(16))) __bf16        v16bf;
typedef __attribute__((ext_vector_type(8)))  float         v8f;
typedef __attribute__((ext_vector_type(4)))  unsigned int  u32x4;
typedef __attribute__((ext_vector_type(4)))  float         f32x4;

__device__ __forceinline__ unsigned short f2bf(float f) {
    unsigned int u = __float_as_uint(f);
    u += 0x7FFFu + ((u >> 16) & 1u);          // round-to-nearest-even
    return (unsigned short)(u >> 16);
}

// ---------------------------------------------------------------------------
// Kernel 1: mod[b][ic] = style[b] . lin_weight[ic] * LIN_SCALE + bias[ic] + 1
// ---------------------------------------------------------------------------
__global__ __launch_bounds__(256)
void k_mod(const float* __restrict__ style,
           const float* __restrict__ lin_w,
           const float* __restrict__ bias,
           float* __restrict__ mod) {
    __shared__ float s_sty[W_DIM];
    const int b   = blockIdx.x;
    const int tid = threadIdx.x;
    for (int i = tid; i < W_DIM; i += 256) s_sty[i] = style[b * W_DIM + i];
    __syncthreads();
    for (int ic = tid; ic < IN_C; ic += 256) {
        const float* lw = lin_w + (size_t)ic * W_DIM;
        float acc = 0.f;
        #pragma unroll 4
        for (int w = 0; w < W_DIM; ++w) acc += s_sty[w] * lw[w];
        mod[b * IN_C + ic] = acc * LIN_SCALE + bias[ic] + 1.0f;
    }
}

// ---------------------------------------------------------------------------
// Kernel 2: modulate + demodulate, write bf16 wmod[b][tap][oc][ic]
//           (ic contiguous => WMMA A-fragment friendly)
// ---------------------------------------------------------------------------
__global__ __launch_bounds__(256)
void k_wmod(const float* __restrict__ weight,
            const float* __restrict__ mod,
            unsigned short* __restrict__ wmod) {
    __shared__ float red[256];
    const int oc  = blockIdx.x;
    const int b   = blockIdx.y;
    const int tid = threadIdx.x;

    float v[2][3];
    float lsum = 0.f;
    #pragma unroll
    for (int i = 0; i < 2; ++i) {
        const int ic = tid + i * 256;
        const float m = mod[b * IN_C + ic] * WEIGHT_SCALE;
        const float* wp = weight + ((size_t)oc * IN_C + ic) * 3;
        #pragma unroll
        for (int k = 0; k < 3; ++k) {
            float t = wp[k] * m;
            v[i][k] = t;
            lsum += t * t;
        }
    }
    red[tid] = lsum;
    __syncthreads();
    for (int s = 128; s > 0; s >>= 1) {
        if (tid < s) red[tid] += red[tid + s];
        __syncthreads();
    }
    const float scale = rsqrtf(red[0] + EPS_DEMOD);
    #pragma unroll
    for (int i = 0; i < 2; ++i) {
        const int ic = tid + i * 256;
        #pragma unroll
        for (int k = 0; k < 3; ++k)
            wmod[(((size_t)b * 3 + k) * OUT_C + oc) * IN_C + ic] = f2bf(v[i][k] * scale);
    }
}

// ---------------------------------------------------------------------------
// Kernel 3: per-sample 1x3 conv as 3 shifted GEMMs via v_wmma_f32_16x16x32_bf16
//   Workgroup: 256 threads = 8 waves, computes a 128(oc) x 256(pos) tile.
//   Wave (wm, wn), wm in 0..1, wn in 0..3, owns 64(oc) x 64(pos):
//     4 M-tiles x 4 N-tiles of 16x16 f32 accumulators (128 VGPRs).
//   Per tap: 8 A-frag + 8 B-frag ds_load_b128 feed 16 WMMAs (1.0 ds/wmma).
// LDS:
//   ldsA[tap][128 oc rows][32 bf16 (+pad)]  rows 80B (20 uints) apart  (30 KB)
//   ldsB[264 pos rows][32 ic bf16 (+pad)]   transposed x tile,
//                                           q = N0-4 .. N0+259          (21 KB)
// ---------------------------------------------------------------------------
__global__ __launch_bounds__(256)
void k_conv_wmma(const float* __restrict__ x,
                 const unsigned short* __restrict__ wmod,
                 float* __restrict__ out) {
    __shared__ unsigned int ldsA[3 * 128 * 20];
    __shared__ unsigned int ldsB[264 * 20];

    const int tid  = threadIdx.x;
    const int lane = tid & 31;
    const int lh   = lane & 15;     // row/col within 16
    const int hs   = lane >> 4;     // K-half selector
    const int wave = tid >> 5;
    const int wm   = wave >> 2;     // 0..1
    const int wn   = wave & 3;      // 0..3

    const int b  = blockIdx.z;
    const int M0 = blockIdx.y * 128;   // oc tile
    const int N0 = blockIdx.x * 256;   // position tile

    const float* xb = x + (size_t)b * IN_C * WID;
    const unsigned short* wb = wmod + (size_t)b * 3 * OUT_C * IN_C;

    v8f acc[4][4];
    #pragma unroll
    for (int m = 0; m < 4; ++m)
        #pragma unroll
        for (int n = 0; n < 4; ++n) {
            v8f z = {0.f, 0.f, 0.f, 0.f, 0.f, 0.f, 0.f, 0.f};
            acc[m][n] = z;
        }

    for (int ics = 0; ics < IN_C; ics += 32) {
        __syncthreads();   // protect previous iteration's fragment reads

        // ---- stage A: 3 taps x [128 oc][32 ic] bf16, 16B chunks, coalesced
        #pragma unroll
        for (int it = 0; it < 6; ++it) {
            const int idx  = tid + it * 256;     // 0..1535
            const int t3   = idx >> 9;           // tap
            const int rem  = idx & 511;
            const int row  = rem >> 2;           // oc row 0..127
            const int quad = rem & 3;            // 16B chunk within 64B row
            const u32x4* src =
                (const u32x4*)(wb + ((size_t)(t3 * OUT_C + M0 + row)) * IN_C + ics);
            *(u32x4*)&ldsA[(t3 * 128 + row) * 20 + quad * 4] = src[quad];
        }

        // ---- stage B: x[ic][q] fp32 -> bf16, transposed to [q][ic]
        //      q covers [N0-4, N0+260): 66 aligned quads per ic row
        unsigned short* ldsBh = (unsigned short*)ldsB;
        #pragma unroll
        for (int it = 0; it < 9; ++it) {
            const int idx = tid + it * 256;
            if (idx < 32 * 66) {
                const int ic = idx / 66;
                const int jq = idx - ic * 66;        // quad index along q
                const int gq = N0 - 4 + jq * 4;      // global position (aligned)
                const float* src = xb + (size_t)(ics + ic) * WID + gq;
                f32x4 v;
                if (gq >= 0 && gq + 4 <= WID) {
                    v = *(const f32x4*)src;
                } else {
                    v.x = (gq + 0 >= 0 && gq + 0 < WID) ? src[0] : 0.f;
                    v.y = (gq + 1 >= 0 && gq + 1 < WID) ? src[1] : 0.f;
                    v.z = (gq + 2 >= 0 && gq + 2 < WID) ? src[2] : 0.f;
                    v.w = (gq + 3 >= 0 && gq + 3 < WID) ? src[3] : 0.f;
                }
                const int r = jq * 4;
                ldsBh[(r + 0) * 40 + ic] = f2bf(v.x);
                ldsBh[(r + 1) * 40 + ic] = f2bf(v.y);
                ldsBh[(r + 2) * 40 + ic] = f2bf(v.z);
                ldsBh[(r + 3) * 40 + ic] = f2bf(v.w);
            }
        }
        __syncthreads();

        // ---- compute: 3 taps x (4 M-tiles x 4 N-tiles) WMMAs
        #pragma unroll
        for (int k = 0; k < 3; ++k) {
            union Frag { u32x4 u[2]; v16bf v; };
            Frag bf[4], af;
            // B fragments: column = output pos, ldsB row shifted by tap (+4 base)
            const int rB = wn * 64 + lh + k + 3;
            #pragma unroll
            for (int nt = 0; nt < 4; ++nt) {
                const int r = (rB + nt * 16) * 20 + hs * 4;
                bf[nt].u[0] = *(const u32x4*)&ldsB[r];
                bf[nt].u[1] = *(const u32x4*)&ldsB[r + 8];
            }
            #pragma unroll
            for (int mt = 0; mt < 4; ++mt) {
                const int rA = (k * 128 + wm * 64 + mt * 16 + lh) * 20;
                af.u[0] = *(const u32x4*)&ldsA[rA + hs * 4];
                af.u[1] = *(const u32x4*)&ldsA[rA + 8 + hs * 4];
                #pragma unroll
                for (int nt = 0; nt < 4; ++nt) {
                    acc[mt][nt] = __builtin_amdgcn_wmma_f32_16x16x32_bf16(
                        false, af.v, false, bf[nt].v, (short)0, acc[mt][nt],
                        false, false);
                }
            }
        }
    }

    // ---- store: C/D layout => VGPR r: lanes 0-15 M=r, lanes 16-31 M=r+8
    float* op = out + (size_t)b * OUT_C * WID;
    const int col0  = N0 + wn * 64 + lh;
    const int row00 = M0 + wm * 64 + hs * 8;
    #pragma unroll
    for (int mt = 0; mt < 4; ++mt) {
        #pragma unroll
        for (int nt = 0; nt < 4; ++nt) {
            const v8f c = acc[mt][nt];
            const int col  = col0 + nt * 16;
            const int rowb = row00 + mt * 16;
            #pragma unroll
            for (int r = 0; r < 8; ++r)
                op[(size_t)(rowb + r) * WID + col] = c[r];
        }
    }
}

// ---------------------------------------------------------------------------
// Launcher
// ---------------------------------------------------------------------------
extern "C" void kernel_launch(void* const* d_in, const int* in_sizes, int n_in,
                              void* d_out, int out_size, void* d_ws, size_t ws_size,
                              hipStream_t stream) {
    const float* x      = (const float*)d_in[0];
    const float* style  = (const float*)d_in[1];
    const float* weight = (const float*)d_in[2];
    const float* lin_w  = (const float*)d_in[3];
    const float* bias   = (const float*)d_in[4];
    float* out = (float*)d_out;

    // workspace: mod f32 [16*512] at 0; wmod bf16 [16*3*512*512] at +64KB
    float* mod = (float*)d_ws;
    unsigned short* wmod = (unsigned short*)((char*)d_ws + 65536);

    k_mod<<<dim3(BATCH), 256, 0, stream>>>(style, lin_w, bias, mod);
    k_wmod<<<dim3(OUT_C, BATCH), 256, 0, stream>>>(weight, mod, wmod);
    k_conv_wmma<<<dim3(WID / 256, OUT_C / 128, BATCH), 256, 0, stream>>>(x, wmod, out);
}
